// TFMRModule_55473797595434
// MI455X (gfx1250) — compile-verified, matrix operand order checked by reading, not run on hardware
//
#include <hip/hip_runtime.h>

typedef __bf16 bf16_t;
typedef __attribute__((ext_vector_type(16))) __bf16 v16bf;
typedef __attribute__((ext_vector_type(8)))  float  v8f;

#define BATCH 2
#define NPTS  2048
#define KNN_  32
#define N1_   1536
#define N2_   921
#define CF    256
#define CC    1024
#define SPTS  ((long)BATCH * NPTS * KNN_)   // 131072 samples in local-feature stage

// ---------------------------------------------------------------------------
// WMMA bf16 GEMM:  D[b][n][m] = sum_k A[b][m][k] * B[b][n][k]  (+ bias[m])
// A: [M][K] bf16 row-major (per-batch stride sA, 0 = shared)
// B: [N][K] bf16 row-major (i.e. B^T of the math B)
// D: f32 and/or bf16, row-major [N][M]
// M%64==0, N%256==0, K%32==0 (guaranteed by callers).
// Block = 256 threads = 8 waves (2 x 4) covering a 64(M) x 256(N) tile.
// Each wave: 32(M) x 64(N) register tile = 2 A-frags x 4 B-frags -> 8 WMMA
// per K-step from 6 fragment loads (fragment reuse for arithmetic intensity).
// ---------------------------------------------------------------------------
union FragU { v16bf v; uint4 q[2]; };

__global__ void __launch_bounds__(256)
k_gemm(const bf16_t* __restrict__ A, const bf16_t* __restrict__ B,
       float* __restrict__ Df, bf16_t* __restrict__ Db,
       const float* __restrict__ bias,
       int M, int N, int K, long sA, long sB, long sD)
{
    int b = blockIdx.z;
    const bf16_t* Ab = A + (long)b * sA;
    const bf16_t* Bb = B + (long)b * sB;
    int wave = threadIdx.x >> 5;
    int lane = threadIdx.x & 31;
    int ot0 = blockIdx.x * 4  + (wave & 1) * 2;   // first 16-tile along M (2 tiles)
    int st0 = blockIdx.y * 16 + (wave >> 1) * 4;  // first 16-tile along N (4 tiles)
    if (ot0 * 16 >= M || st0 * 16 >= N) return;   // wave-uniform
    int row = lane & 15;
    int kb  = (lane < 16) ? 0 : 8;
    const bf16_t* ap = Ab + (long)(ot0 * 16 + row) * K + kb;
    const bf16_t* bp = Bb + (long)(st0 * 16 + row) * K + kb;
    long aStride = (long)16 * K;   // one 16-row tile of A
    long bStride = (long)16 * K;   // one 16-row tile of B
    v8f acc[8];                    // [mi*4 + ni]
#pragma unroll
    for (int i = 0; i < 8; ++i)
#pragma unroll
        for (int r = 0; r < 8; ++r) acc[i][r] = 0.0f;

    for (int k0 = 0; k0 < K; k0 += 32) {
        FragU fa[2], fb[4];
#pragma unroll
        for (int mi = 0; mi < 2; ++mi) {
            const bf16_t* p = ap + (long)mi * aStride + k0;
            fa[mi].q[0] = *(const uint4*)(p);
            fa[mi].q[1] = *(const uint4*)(p + 16);
        }
#pragma unroll
        for (int ni = 0; ni < 4; ++ni) {
            const bf16_t* p = bp + (long)ni * bStride + k0;
            fb[ni].q[0] = *(const uint4*)(p);
            fb[ni].q[1] = *(const uint4*)(p + 16);
        }
#pragma unroll
        for (int mi = 0; mi < 2; ++mi)
#pragma unroll
            for (int ni = 0; ni < 4; ++ni)
                acc[mi*4+ni] = __builtin_amdgcn_wmma_f32_16x16x32_bf16(
                    false, fa[mi].v, false, fb[ni].v,
                    (short)0, acc[mi*4+ni], false, false);
    }

#pragma unroll
    for (int mi = 0; mi < 2; ++mi) {
        int mb = (ot0 + mi) * 16 + ((lane < 16) ? 0 : 8);
#pragma unroll
        for (int ni = 0; ni < 4; ++ni) {
            int n = (st0 + ni) * 16 + row;
            long dofs = (long)b * sD + (long)n * M + mb;
            float vals[8];
#pragma unroll
            for (int r = 0; r < 8; ++r) vals[r] = acc[mi*4+ni][r];
            if (bias) {
#pragma unroll
                for (int r = 0; r < 8; ++r) vals[r] += bias[mb + r];
            }
            if (Df) {
#pragma unroll
                for (int r = 0; r < 8; ++r) Df[dofs + r] = vals[r];
            }
            if (Db) {
#pragma unroll
                for (int r = 0; r < 8; ++r) Db[dofs + r] = (bf16_t)vals[r];
            }
        }
    }
}

// ------------------------- small utility kernels ---------------------------
__global__ void __launch_bounds__(256)
k_f2b(const float* __restrict__ s, bf16_t* __restrict__ d, long n)
{
    long t = (long)blockIdx.x * 256 + threadIdx.x;
    if (t < n) d[t] = (bf16_t)s[t];
}

__global__ void __launch_bounds__(64)
k_zero(float* p, int n)
{
    int t = threadIdx.x;
    if (t < n) p[t] = 0.f;
}

// ball query: first KNN_ in-radius indices in ascending index order, pad w/ first
__global__ void __launch_bounds__(256)
k_ball(const float* __restrict__ xyz, int* __restrict__ inds)
{
    long t = (long)blockIdx.x * 256 + threadIdx.x;
    if (t >= (long)BATCH * NPTS) return;
    int b = (int)(t / NPTS), n = (int)(t % NPTS);
    const float* base = xyz + (long)b * NPTS * 3;
    float cx = base[n*3+0], cy = base[n*3+1], cz = base[n*3+2];
    int* out = inds + t * KNN_;
    int cnt = 0;
    for (int j = 0; j < NPTS && cnt < KNN_; ++j) {
        float dx = base[j*3+0]-cx, dy = base[j*3+1]-cy, dz = base[j*3+2]-cz;
        float d2 = dx*dx + dy*dy + dz*dz;
        if (sqrtf(d2) <= 0.2f) out[cnt++] = j;
    }
    int first = (cnt > 0) ? out[0] : n;
    for (int k = cnt; k < KNN_; ++k) out[k] = first;
}

// layer1: out[s][o] = W1[o][0:6] . [ctr, nbr-ctr]   (K=6, VALU)
__global__ void __launch_bounds__(256)
k_layer1(const float* __restrict__ xyz, const int* __restrict__ inds,
         const float* __restrict__ W1, float* __restrict__ out)
{
    long t = (long)blockIdx.x * 256 + threadIdx.x;
    if (t >= SPTS * 128) return;
    long s = t >> 7; int o = (int)(t & 127);
    long bn = s >> 5;                  // b*NPTS + n
    long b  = bn / NPTS;
    const float* ctr = xyz + bn * 3;
    int j = inds[s];
    const float* pj = xyz + (b * NPTS + j) * 3;
    float f0 = ctr[0], f1 = ctr[1], f2 = ctr[2];
    float f3 = pj[0]-f0, f4 = pj[1]-f1, f5 = pj[2]-f2;
    const float* w = W1 + o * 6;
    out[t] = w[0]*f0 + w[1]*f1 + w[2]*f2 + w[3]*f3 + w[4]*f4 + w[5]*f5;
}

// group-norm stats: sum & sumsq per (batch, group) via block reduce + atomics
__global__ void __launch_bounds__(256)
k_gnstat(const float* __restrict__ x, float* __restrict__ stats,
         int C, int gc, long samplesPerB)
{
    int b = blockIdx.z, g = blockIdx.y, chunk = blockIdx.x, nchunks = gridDim.x;
    long per = samplesPerB / nchunks;
    long s0  = (long)chunk * per;
    long total = per * gc;
    const float* xb = x + (long)b * samplesPerB * C;
    float s1 = 0.f, s2 = 0.f;
    for (long e = threadIdx.x; e < total; e += 256) {
        long s = s0 + e / gc;
        int  c = g * gc + (int)(e % gc);
        float v = xb[s * C + c];
        s1 += v; s2 += v * v;
    }
    __shared__ float r1[256], r2[256];
    r1[threadIdx.x] = s1; r2[threadIdx.x] = s2; __syncthreads();
    for (int w = 128; w > 0; w >>= 1) {
        if ((int)threadIdx.x < w) { r1[threadIdx.x] += r1[threadIdx.x+w]; r2[threadIdx.x] += r2[threadIdx.x+w]; }
        __syncthreads();
    }
    if (threadIdx.x == 0) {
        int G = gridDim.y;
        atomicAdd(&stats[((long)b*G + g)*2 + 0], r1[0]);
        atomicAdd(&stats[((long)b*G + g)*2 + 1], r2[0]);
    }
}

// group-norm apply + activation (0 none, 1 relu, 2 leaky 0.2); dual store
__global__ void __launch_bounds__(256)
k_gn_act(const float* __restrict__ x, const float* __restrict__ stats,
         const float* __restrict__ gamma, const float* __restrict__ beta,
         float* __restrict__ of, bf16_t* __restrict__ ob,
         int C, int gc, long samplesPerB, float cntInv, int act, long total)
{
    long t = (long)blockIdx.x * 256 + threadIdx.x;
    if (t >= total) return;
    long s = t / C; int c = (int)(t % C);
    int b = (int)(s / samplesPerB);
    int g = c / gc; int G = C / gc;
    float m  = stats[((long)b*G + g)*2 + 0] * cntInv;
    float vv = stats[((long)b*G + g)*2 + 1] * cntInv - m * m;
    float v = (x[t] - m) * rsqrtf(vv + 1e-5f) * gamma[c] + beta[c];
    if (act == 1)      v = fmaxf(v, 0.f);
    else if (act == 2) v = (v >= 0.f) ? v : 0.2f * v;
    if (of) of[t] = v;
    if (ob) ob[t] = (bf16_t)v;
}

// attention-block tail: x = x + relu(gn(tout)); also fill concat slot
__global__ void __launch_bounds__(256)
k_gn_res(const float* __restrict__ tout, const float* __restrict__ stats,
         const float* __restrict__ gamma, const float* __restrict__ beta,
         float* __restrict__ xcur, bf16_t* __restrict__ xcurb,
         bf16_t* __restrict__ featcat, int slot)
{
    long t = (long)blockIdx.x * 256 + threadIdx.x;
    if (t >= (long)BATCH * NPTS * CF) return;
    long s = t / CF; int c = (int)(t % CF);
    int b = (int)(s / NPTS);
    int g = c >> 5;
    const float cntInv = 1.f / (32.f * (float)NPTS);
    float m  = stats[((long)b*8 + g)*2 + 0] * cntInv;
    float vv = stats[((long)b*8 + g)*2 + 1] * cntInv - m * m;
    float v = (tout[t] - m) * rsqrtf(vv + 1e-5f) * gamma[c] + beta[c];
    v = fmaxf(v, 0.f);
    float nx = xcur[t] + v;
    xcur[t]  = nx;
    xcurb[t] = (bf16_t)nx;
    featcat[s * (long)CC + slot * CF + c] = (bf16_t)nx;
}

// max over the KNN axis of [bn][k][C] -> [bn][C]
__global__ void __launch_bounds__(256)
k_maxk(const float* __restrict__ x, float* __restrict__ xcur, bf16_t* __restrict__ xcurb)
{
    long t = (long)blockIdx.x * 256 + threadIdx.x;
    if (t >= (long)BATCH * NPTS * CF) return;
    long bn = t / CF; int c = (int)(t % CF);
    const float* p = x + bn * KNN_ * (long)CF + c;
    float m = p[0];
    for (int k = 1; k < KNN_; ++k) m = fmaxf(m, p[(long)k * CF]);
    xcur[t] = m; xcurb[t] = (bf16_t)m;
}

// row softmax in place (row = blockIdx.x)
__global__ void __launch_bounds__(256)
k_softmax(float* __restrict__ att, int C)
{
    long row = blockIdx.x;
    float* p = att + row * (long)C;
    __shared__ float red[256];
    int tid = threadIdx.x;
    float m = -3.4e38f;
    for (int j = tid; j < C; j += 256) m = fmaxf(m, p[j]);
    red[tid] = m; __syncthreads();
    for (int w = 128; w > 0; w >>= 1) { if (tid < w) red[tid] = fmaxf(red[tid], red[tid+w]); __syncthreads(); }
    m = red[0]; __syncthreads();
    float s = 0.f;
    for (int j = tid; j < C; j += 256) { float e = __expf(p[j] - m); p[j] = e; s += e; }
    red[tid] = s; __syncthreads();
    for (int w = 128; w > 0; w >>= 1) { if (tid < w) red[tid] += red[tid+w]; __syncthreads(); }
    float inv = 1.f / red[0];
    for (int j = tid; j < C; j += 256) p[j] *= inv;
}

__global__ void __launch_bounds__(256)
k_colsum(const float* __restrict__ att, float* __restrict__ colsum)
{
    int b = blockIdx.y;
    int m = blockIdx.x * 256 + threadIdx.x;
    if (m >= NPTS) return;
    const float* p = att + (long)b * NPTS * NPTS + m;
    float s = 0.f;
    for (int n = 0; n < NPTS; ++n) s += p[(long)n * NPTS];
    colsum[(long)b * NPTS + m] = s;
}

// attT[b][m][n] = bf16( att[b][n][m] / (1e-9 + colsum[b][m]) )
__global__ void __launch_bounds__(256)
k_attT(const float* __restrict__ att, const float* __restrict__ colsum,
       bf16_t* __restrict__ attT)
{
    long t = (long)blockIdx.x * 256 + threadIdx.x;
    if (t >= (long)BATCH * NPTS * NPTS) return;
    long b = t / ((long)NPTS * NPTS);
    long r = t % ((long)NPTS * NPTS);
    int m = (int)(r / NPTS), n = (int)(r % NPTS);
    float cs = colsum[b * NPTS + m];
    float v  = att[(b * NPTS + n) * NPTS + m] / (1e-9f + cs);
    attT[t] = (bf16_t)v;
}

// xv channel-major bf16: xvcm[b][c][n] = xvT[b][n][c]
__global__ void __launch_bounds__(256)
k_xvtrans(const float* __restrict__ xvT, bf16_t* __restrict__ xvcm)
{
    long t = (long)blockIdx.x * 256 + threadIdx.x;
    if (t >= (long)BATCH * NPTS * CF) return;
    long s = t / CF; int c = (int)(t % CF);
    int b = (int)(s / NPTS), n = (int)(s % NPTS);
    xvcm[((long)b * CF + c) * NPTS + n] = (bf16_t)xvT[t];
}

// diff[b][n][c] = bf16( x[b][n][c] - xr_cm[b][c][n] )
__global__ void __launch_bounds__(256)
k_diff(const float* __restrict__ xcur, const float* __restrict__ xrcm,
       bf16_t* __restrict__ diffb)
{
    long t = (long)blockIdx.x * 256 + threadIdx.x;
    if (t >= (long)BATCH * NPTS * CF) return;
    long s = t / CF; int c = (int)(t % CF);
    int b = (int)(s / NPTS), n = (int)(s % NPTS);
    float v = xcur[t] - xrcm[((long)b * CF + c) * NPTS + n];
    diffb[t] = (bf16_t)v;
}

// L2-normalize each row of [rows][C] f32 -> bf16
__global__ void __launch_bounds__(256)
k_rownorm(const float* __restrict__ x, bf16_t* __restrict__ out, int C)
{
    long row = blockIdx.x;
    const float* p = x + row * (long)C;
    __shared__ float red[256];
    int tid = threadIdx.x;
    float s = 0.f;
    for (int j = tid; j < C; j += 256) { float v = p[j]; s += v * v; }
    red[tid] = s; __syncthreads();
    for (int w = 128; w > 0; w >>= 1) { if (tid < w) red[tid] += red[tid+w]; __syncthreads(); }
    float inv = 1.f / (sqrtf(red[0]) + 1e-8f);
    for (int j = tid; j < C; j += 256) out[row * (long)C + j] = (bf16_t)(p[j] * inv);
}

// stable descending rank-sort: order[rank]=i, sorted[rank]=score[i]
__global__ void __launch_bounds__(256)
k_rank(const float* __restrict__ score, int n, int* __restrict__ order,
       float* __restrict__ sorted)
{
    int b = blockIdx.y;
    int i = blockIdx.x * 256 + threadIdx.x;
    if (i >= n) return;
    const float* s = score + (long)b * n;
    float si = s[i];
    int r = 0;
    for (int j = 0; j < n; ++j) {
        float sj = s[j];
        r += (sj > si) || (sj == si && j < i);
    }
    order[(long)b * n + r] = i;
    if (sorted) sorted[(long)b * n + r] = si;
}

__global__ void __launch_bounds__(256)
k_gather_feat(const bf16_t* __restrict__ f, const int* __restrict__ order,
              bf16_t* __restrict__ out)
{
    long t = (long)blockIdx.x * 256 + threadIdx.x;
    if (t >= (long)BATCH * N1_ * CC) return;
    long br = t / CC; int c = (int)(t % CC);
    int b = (int)(br / N1_), r = (int)(br % N1_);
    int idx = order[(long)b * NPTS + r];
    out[t] = f[((long)b * NPTS + idx) * CC + c];
}

__global__ void __launch_bounds__(256)
k_gather_pts(const float* __restrict__ pts, const int* __restrict__ order,
             float* __restrict__ out)
{
    long t = (long)blockIdx.x * 256 + threadIdx.x;
    if (t >= (long)BATCH * N1_) return;
    int b = (int)(t / N1_), r = (int)(t % N1_);
    int idx = order[(long)b * NPTS + r];
    const float* p = pts + ((long)b * NPTS + idx) * 3;
    out[t*3+0] = p[0]; out[t*3+1] = p[1]; out[t*3+2] = p[2];
}

__global__ void __launch_bounds__(256)
k_rowmax(const float* __restrict__ x, float* __restrict__ out, int C)
{
    long row = blockIdx.x;
    const float* p = x + row * (long)C;
    __shared__ float red[256];
    int tid = threadIdx.x;
    float m = -3.4e38f;
    for (int j = tid; j < C; j += 256) m = fmaxf(m, p[j]);
    red[tid] = m; __syncthreads();
    for (int w = 128; w > 0; w >>= 1) { if (tid < w) red[tid] = fmaxf(red[tid], red[tid+w]); __syncthreads(); }
    if (tid == 0) out[row] = red[0];
}

// final: top-3 correspondence + output packing
__global__ void __launch_bounds__(256)
k_final(const float* __restrict__ sim, const int* __restrict__ smorder,
        const float* __restrict__ xsorted, const float* __restrict__ srcsg,
        const float* __restrict__ tgtsg, float* __restrict__ out)
{
    long t = (long)blockIdx.x * 256 + threadIdx.x;
    if (t >= (long)BATCH * N2_) return;
    int b = (int)(t / N2_), r = (int)(t % N2_);
    int i = smorder[(long)b * N1_ + r];
    const float* row = sim + ((long)b * N1_ + i) * N1_;
    float v1 = -3.4e38f, v2 = -3.4e38f, v3 = -3.4e38f;
    int i1 = 0, i2 = 0, i3 = 0;
    for (int m = 0; m < N1_; ++m) {
        float v = row[m];
        if (v > v1)      { v3 = v2; i3 = i2; v2 = v1; i2 = i1; v1 = v; i1 = m; }
        else if (v > v2) { v3 = v2; i3 = i2; v2 = v; i2 = m; }
        else if (v > v3) { v3 = v; i3 = m; }
    }
    float inv = 1.f / (v1 + v2 + v3 + 1e-8f);
    float w1 = v1 * inv, w2 = v2 * inv, w3 = v3 * inv;
    float* o_src  = out;
    float* o_corr = out + (long)BATCH * N2_ * 3;
    float* o_xol  = o_corr + (long)BATCH * N2_ * 3;
    float* o_idx  = o_xol + (long)BATCH * N2_;
    const float* sp = srcsg + ((long)b * N1_ + i) * 3;
    o_src[t*3+0] = sp[0]; o_src[t*3+1] = sp[1]; o_src[t*3+2] = sp[2];
    const float* t1 = tgtsg + ((long)b * N1_ + i1) * 3;
    const float* t2 = tgtsg + ((long)b * N1_ + i2) * 3;
    const float* t3 = tgtsg + ((long)b * N1_ + i3) * 3;
#pragma unroll
    for (int c = 0; c < 3; ++c)
        o_corr[t*3+c] = w1 * t1[c] + w2 * t2[c] + w3 * t3[c];
    o_xol[t] = xsorted[(long)b * NPTS + i];
    o_idx[t] = (float)i;
}

// ===========================================================================
extern "C" void kernel_launch(void* const* d_in, const int* in_sizes, int n_in,
                              void* d_out, int out_size, void* d_ws, size_t ws_size,
                              hipStream_t stream)
{
    (void)in_sizes; (void)n_in; (void)out_size; (void)ws_size;
    const float* src   = (const float*)d_in[0];
    const float* tgt   = (const float*)d_in[1];
    const float* xol   = (const float*)d_in[2];
    const float* yol   = (const float*)d_in[3];
    const float* lf_w1 = (const float*)d_in[4];
    const float* lf_g1 = (const float*)d_in[5];
    const float* lf_b1 = (const float*)d_in[6];
    const float* lf_w2 = (const float*)d_in[7];
    const float* lf_g2 = (const float*)d_in[8];
    const float* lf_b2 = (const float*)d_in[9];
    const float* lf_w3 = (const float*)d_in[10];
    const float* lf_g3 = (const float*)d_in[11];
    const float* lf_b3 = (const float*)d_in[12];
    const float* qk_w  = (const float*)d_in[13];
    const float* v_w   = (const float*)d_in[14];
    const float* v_b   = (const float*)d_in[15];
    const float* t_w   = (const float*)d_in[16];
    const float* t_b   = (const float*)d_in[17];
    const float* gn_g  = (const float*)d_in[18];
    const float* gn_b  = (const float*)d_in[19];
    const float* fu_w  = (const float*)d_in[20];
    const float* fu_g  = (const float*)d_in[21];
    const float* fu_b  = (const float*)d_in[22];
    float* out = (float*)d_out;

    char* ws = (char*)d_ws;
    size_t cur = 0;
    auto alloc = [&](size_t bytes) -> void* {
        void* p = ws + cur;
        cur = (cur + bytes + 255) & ~(size_t)255;
        return p;
    };

    // bf16 weights (persistent)
    bf16_t* w2b    = (bf16_t*)alloc(256*128*2);
    bf16_t* w3b    = (bf16_t*)alloc(256*256*2);
    bf16_t* qkwb   = (bf16_t*)alloc(4*64*256*2);
    bf16_t* vwb    = (bf16_t*)alloc((size_t)4*256*256*2);
    bf16_t* twb    = (bf16_t*)alloc((size_t)4*256*256*2);
    bf16_t* fusewb = (bf16_t*)alloc((size_t)1024*1024*2);
    // embed scratch
    int*    inds    = (int*)alloc((size_t)SPTS*4);
    float*  buff32  = (float*)alloc((size_t)SPTS*256*4);
    bf16_t* actb    = (bf16_t*)alloc((size_t)SPTS*256*2);
    float*  xcur    = (float*)alloc((size_t)BATCH*NPTS*CF*4);
    bf16_t* xcurb   = (bf16_t*)alloc((size_t)BATCH*NPTS*CF*2);
    bf16_t* qkb     = (bf16_t*)alloc((size_t)BATCH*NPTS*64*2);
    float*  xvT     = (float*)alloc((size_t)BATCH*NPTS*CF*4);
    bf16_t* xvcmb   = (bf16_t*)alloc((size_t)BATCH*CF*NPTS*2);
    float*  attf    = (float*)alloc((size_t)BATCH*NPTS*NPTS*4);
    float*  colsum  = (float*)alloc((size_t)BATCH*NPTS*4);
    bf16_t* attTb   = (bf16_t*)alloc((size_t)BATCH*NPTS*NPTS*2);
    float*  xrcm    = (float*)alloc((size_t)BATCH*CF*NPTS*4);
    bf16_t* diffb   = (bf16_t*)alloc((size_t)BATCH*NPTS*CF*2);
    float*  tout    = (float*)alloc((size_t)BATCH*NPTS*CF*4);
    bf16_t* featcat = (bf16_t*)alloc((size_t)BATCH*NPTS*CC*2);
    float*  fuseout = (float*)alloc((size_t)BATCH*NPTS*CC*4);
    bf16_t* fxb_src = (bf16_t*)alloc((size_t)BATCH*NPTS*CC*2);
    bf16_t* fxb_tgt = (bf16_t*)alloc((size_t)BATCH*NPTS*CC*2);
    int*    orderx  = (int*)alloc((size_t)BATCH*NPTS*4);
    float*  xsorted = (float*)alloc((size_t)BATCH*NPTS*4);
    int*    ordery  = (int*)alloc((size_t)BATCH*NPTS*4);
    bf16_t* fxg     = (bf16_t*)alloc((size_t)BATCH*N1_*CC*2);
    bf16_t* fyg     = (bf16_t*)alloc((size_t)BATCH*N1_*CC*2);
    float*  srcsg   = (float*)alloc((size_t)BATCH*N1_*3*4);
    float*  tgtsg   = (float*)alloc((size_t)BATCH*N1_*3*4);
    float*  sim     = (float*)alloc((size_t)BATCH*N1_*N1_*4);
    float*  smax    = (float*)alloc((size_t)BATCH*N1_*4);
    int*    smorder = (int*)alloc((size_t)BATCH*N1_*4);
    float*  stats   = (float*)alloc(64*4);

    auto g1 = [](long n) { return dim3((unsigned)((n + 255) / 256)); };

    // ---- weight conversion (once) ----
    k_f2b<<<g1(256*128),       256, 0, stream>>>(lf_w2, w2b,    256*128);
    k_f2b<<<g1(256*256),       256, 0, stream>>>(lf_w3, w3b,    256*256);
    k_f2b<<<g1(4*64*256),      256, 0, stream>>>(qk_w,  qkwb,   4*64*256);
    k_f2b<<<g1(4L*256*256),    256, 0, stream>>>(v_w,   vwb,    4L*256*256);
    k_f2b<<<g1(4L*256*256),    256, 0, stream>>>(t_w,   twb,    4L*256*256);
    k_f2b<<<g1(1024L*1024),    256, 0, stream>>>(fu_w,  fusewb, 1024L*1024);

    auto run_embed = [&](const float* xyz, bf16_t* fxb_out) {
        // ---- local features ----
        k_ball<<<g1((long)BATCH*NPTS), 256, 0, stream>>>(xyz, inds);
        k_layer1<<<g1(SPTS*128), 256, 0, stream>>>(xyz, inds, lf_w1, buff32);
        // gn1 + relu -> act1b (bf16 [S][128])
        k_zero<<<1, 64, 0, stream>>>(stats, 64);
        k_gnstat<<<dim3(64, 4, BATCH), 256, 0, stream>>>(buff32, stats, 128, 32, NPTS*(long)KNN_);
        k_gn_act<<<g1(SPTS*128), 256, 0, stream>>>(buff32, stats, lf_g1, lf_b1,
            nullptr, actb, 128, 32, NPTS*(long)KNN_, 1.f/(32.f*NPTS*KNN_), 1, SPTS*128);
        // layer2 GEMM: [256x128] x [S][128] -> buff32 [S][256]
        k_gemm<<<dim3(4, 512, 1), 256, 0, stream>>>(w2b, actb, buff32, nullptr, nullptr,
            256, (int)SPTS, 128, 0, 0, 0);
        k_zero<<<1, 64, 0, stream>>>(stats, 64);
        k_gnstat<<<dim3(64, 8, BATCH), 256, 0, stream>>>(buff32, stats, 256, 32, NPTS*(long)KNN_);
        k_gn_act<<<g1(SPTS*256), 256, 0, stream>>>(buff32, stats, lf_g2, lf_b2,
            nullptr, actb, 256, 32, NPTS*(long)KNN_, 1.f/(32.f*NPTS*KNN_), 1, SPTS*256);
        // layer3 GEMM
        k_gemm<<<dim3(4, 512, 1), 256, 0, stream>>>(w3b, actb, buff32, nullptr, nullptr,
            256, (int)SPTS, 256, 0, 0, 0);
        k_zero<<<1, 64, 0, stream>>>(stats, 64);
        k_gnstat<<<dim3(64, 8, BATCH), 256, 0, stream>>>(buff32, stats, 256, 32, NPTS*(long)KNN_);
        k_gn_act<<<g1(SPTS*256), 256, 0, stream>>>(buff32, stats, lf_g3, lf_b3,
            buff32, nullptr, 256, 32, NPTS*(long)KNN_, 1.f/(32.f*NPTS*KNN_), 1, SPTS*256);
        // max over k -> x (f32 + bf16)
        k_maxk<<<g1((long)BATCH*NPTS*CF), 256, 0, stream>>>(buff32, xcur, xcurb);

        // ---- 4 attention blocks ----
        for (int i = 0; i < 4; ++i) {
            const bf16_t* qkw = qkwb + (size_t)i * 64 * 256;
            const bf16_t* vw  = vwb  + (size_t)i * 256 * 256;
            const bf16_t* tw  = twb  + (size_t)i * 256 * 256;
            // qk: [64x256] x [n][256] -> qkb [b][n][64]
            k_gemm<<<dim3(1, 8, BATCH), 256, 0, stream>>>(qkw, xcurb, nullptr, qkb, nullptr,
                64, NPTS, 256, 0, (long)NPTS*256, (long)NPTS*64);
            // att Gram: [n][64] x [m][64] -> attf [b][n][m]
            k_gemm<<<dim3(32, 8, BATCH), 256, 0, stream>>>(qkb, qkb, attf, nullptr, nullptr,
                NPTS, NPTS, 64, (long)NPTS*64, (long)NPTS*64, (long)NPTS*NPTS);
            k_softmax<<<dim3(BATCH*NPTS), 256, 0, stream>>>(attf, NPTS);
            k_colsum<<<dim3(NPTS/256, BATCH), 256, 0, stream>>>(attf, colsum);
            k_attT<<<g1((long)BATCH*NPTS*NPTS), 256, 0, stream>>>(attf, colsum, attTb);
            // xv: [256x256] x [n][256] + bias -> xvT f32
            k_gemm<<<dim3(4, 8, BATCH), 256, 0, stream>>>(vw, xcurb, xvT, nullptr, v_b + i*256,
                256, NPTS, 256, 0, (long)NPTS*256, (long)NPTS*256);
            k_xvtrans<<<g1((long)BATCH*NPTS*CF), 256, 0, stream>>>(xvT, xvcmb);
            // xr: attT [m][n] x xvcm [c][n] -> xrcm [b][c][m]
            k_gemm<<<dim3(32, 1, BATCH), 256, 0, stream>>>(attTb, xvcmb, xrcm, nullptr, nullptr,
                NPTS, 256, NPTS, (long)NPTS*NPTS, (long)CF*NPTS, (long)CF*NPTS);
            k_diff<<<g1((long)BATCH*NPTS*CF), 256, 0, stream>>>(xcur, xrcm, diffb);
            // t: [256x256] x diff + bias -> tout
            k_gemm<<<dim3(4, 8, BATCH), 256, 0, stream>>>(tw, diffb, tout, nullptr, t_b + i*256,
                256, NPTS, 256, 0, (long)NPTS*256, (long)NPTS*256);
            k_zero<<<1, 64, 0, stream>>>(stats, 64);
            k_gnstat<<<dim3(8, 8, BATCH), 256, 0, stream>>>(tout, stats, 256, 32, NPTS);
            k_gn_res<<<g1((long)BATCH*NPTS*CF), 256, 0, stream>>>(tout, stats,
                gn_g + i*256, gn_b + i*256, xcur, xcurb, featcat, i);
        }
        // ---- fuse + GN(16) + leaky + L2 normalize ----
        k_gemm<<<dim3(16, 8, BATCH), 256, 0, stream>>>(fusewb, featcat, fuseout, nullptr, nullptr,
            1024, NPTS, 1024, 0, (long)NPTS*CC, (long)NPTS*CC);
        k_zero<<<1, 64, 0, stream>>>(stats, 64);
        k_gnstat<<<dim3(8, 16, BATCH), 256, 0, stream>>>(fuseout, stats, 1024, 64, NPTS);
        k_gn_act<<<g1((long)BATCH*NPTS*CC), 256, 0, stream>>>(fuseout, stats, fu_g, fu_b,
            fuseout, nullptr, 1024, 64, NPTS, 1.f/(64.f*NPTS), 2, (long)BATCH*NPTS*CC);
        k_rownorm<<<dim3(BATCH*NPTS), 256, 0, stream>>>(fuseout, fxb_out, CC);
    };

    run_embed(src, fxb_src);
    run_embed(tgt, fxb_tgt);

    // ---- ordering & gathers ----
    k_rank<<<dim3(NPTS/256, BATCH), 256, 0, stream>>>(xol, NPTS, orderx, xsorted);
    k_rank<<<dim3(NPTS/256, BATCH), 256, 0, stream>>>(yol, NPTS, ordery, nullptr);
    k_gather_feat<<<g1((long)BATCH*N1_*CC), 256, 0, stream>>>(fxb_src, orderx, fxg);
    k_gather_feat<<<g1((long)BATCH*N1_*CC), 256, 0, stream>>>(fxb_tgt, ordery, fyg);
    k_gather_pts<<<g1((long)BATCH*N1_), 256, 0, stream>>>(src, orderx, srcsg);
    k_gather_pts<<<g1((long)BATCH*N1_), 256, 0, stream>>>(tgt, ordery, tgtsg);

    // ---- similarity GEMM: sim[b][n][m] = f_x[n] . f_y[m] ----
    k_gemm<<<dim3(24, 6, BATCH), 256, 0, stream>>>(fyg, fxg, sim, nullptr, nullptr,
        N1_, N1_, CC, (long)N1_*CC, (long)N1_*CC, (long)N1_*N1_);

    // ---- sim-max ranking + final correspondence ----
    k_rowmax<<<dim3(BATCH*N1_), 256, 0, stream>>>(sim, smax, N1_);
    k_rank<<<dim3((N1_+255)/256, BATCH), 256, 0, stream>>>(smax, N1_, smorder, nullptr);
    k_final<<<g1((long)BATCH*N2_), 256, 0, stream>>>(sim, smorder, xsorted, srcsg, tgtsg, out);
}
